// IPMPUpdateLayer_18897856103195
// MI455X (gfx1250) — compile-verified, matrix operand order checked by reading, not run on hardware
//
#include <hip/hip_runtime.h>
#include <stdint.h>

#define N_NODES 10000
#define N_EDGES 300000
#define CS 256
#define CZ 128
#define CH 256
#define PP 8
#define DMSG 672   // 2*CS + CZ + 4*P
#define DEH 384    // CS + CZ

typedef unsigned int  u32;
typedef unsigned short u16;

typedef __attribute__((ext_vector_type(16))) __bf16 v16bf;
typedef __attribute__((ext_vector_type(8)))  float  v8f;

union FragAB { uint4 q[2]; v16bf v; };

// ---------------- helpers ----------------

__device__ __forceinline__ u16 f2bf(float f) {
  union { float f; u32 u; } v; v.f = f;
  u32 r = 0x7FFFu + ((v.u >> 16) & 1u);
  return (u16)((v.u + r) >> 16);
}
__device__ __forceinline__ u32 pack2bf(float a, float b) {
  return (u32)f2bf(a) | ((u32)f2bf(b) << 16);
}

__device__ __forceinline__ v8f wmma_bf16(v16bf a, v16bf b, v8f c) {
  // D = A(16xK=32, bf16) x B(32x16, bf16) + C(16x16, f32)
  return __builtin_amdgcn_wmma_f32_16x16x32_bf16(false, a, false, b, (short)0, c,
                                                 false, false);
}

// A fragment from an LDS tile: row = lane&15, K halves selected by lane>>4.
// dword layout within a row: consecutive K pairs; two ds_load_b128 per frag.
__device__ __forceinline__ v16bf lds_a_frag(const u32* base, int stride_dw, int kt,
                                            int lane) {
  int m = lane & 15, h = lane >> 4;
  FragAB f;
  const u32* r = base + m * stride_dw + kt * 16 + h * 4;
  f.q[0] = *reinterpret_cast<const uint4*>(r);
  f.q[1] = *reinterpret_cast<const uint4*>(r + 8);
  return f.v;
}

// B fragment from packed global weights: [kt][nt][lane][8 dwords]
__device__ __forceinline__ v16bf glb_b_frag(const u32* Wp, int NT, int kt, int nt,
                                            int lane) {
  FragAB f;
  const uint4* q =
      reinterpret_cast<const uint4*>(Wp) + ((size_t)(kt * NT + nt) * 32 + lane) * 2;
  f.q[0] = q[0];
  f.q[1] = q[1];
  return f.v;
}

// ---------------- utility kernels ----------------

__global__ void k_zero_f32(float* __restrict__ p, long long n) {
  long long i = (long long)blockIdx.x * blockDim.x + threadIdx.x;
  long long st = (long long)gridDim.x * blockDim.x;
  for (; i < n; i += st) p[i] = 0.0f;
}

// Pack row-major f32 W[K,N] into bf16 B-fragment layout [KT][NT][32][16 bf16].
__global__ void k_pack_b(const float* __restrict__ W, int K, int Ncols,
                         u32* __restrict__ out) {
  int KT = K >> 5, NT = Ncols >> 4;
  int total = KT * NT * 32 * 8;  // dwords
  for (int idx = blockIdx.x * blockDim.x + threadIdx.x; idx < total;
       idx += gridDim.x * blockDim.x) {
    int j = idx & 7;
    int lane = (idx >> 3) & 31;
    int frag = idx >> 8;
    int nt = frag % NT;
    int kt = frag / NT;
    int h = lane >> 4;
    int n = (nt << 4) + (lane & 15);
    int k = (kt << 5) + ((j >> 2) << 4) + (h << 3) + ((j & 3) << 1);
    out[idx] = pack2bf(W[(size_t)k * Ncols + n], W[(size_t)(k + 1) * Ncols + n]);
  }
}

// ---------------- K0: node prep (points + bf16 copy of s) ----------------

__global__ __launch_bounds__(256) void k0_node_prep(
    const float* __restrict__ s, const float* __restrict__ rot,
    const float* __restrict__ trans, const float* __restrict__ Wp,
    const float* __restrict__ bp, u16* __restrict__ sbf, float* __restrict__ ptsg) {
  int wv = threadIdx.x >> 5, lane = threadIdx.x & 31;
  int n = blockIdx.x * 8 + wv;
  if (n >= N_NODES) return;
  const float* srow = s + (size_t)n * CS;
  u32* sb = reinterpret_cast<u32*>(sbf + (size_t)n * CS);
  for (int d = lane; d < CS / 2; d += 32)
    sb[d] = pack2bf(srow[2 * d], srow[2 * d + 1]);
  if (lane < PP) {
    int p = lane;
    float l0 = bp[p * 3 + 0], l1 = bp[p * 3 + 1], l2 = bp[p * 3 + 2];
    for (int k = 0; k < CS; ++k) {
      float sv = srow[k];
      l0 += sv * Wp[k * 24 + p * 3 + 0];
      l1 += sv * Wp[k * 24 + p * 3 + 1];
      l2 += sv * Wp[k * 24 + p * 3 + 2];
    }
    const float* R = rot + (size_t)n * 9;
    const float* T = trans + (size_t)n * 3;
    ptsg[(size_t)n * 24 + p * 3 + 0] = R[0] * l0 + R[1] * l1 + R[2] * l2 + T[0];
    ptsg[(size_t)n * 24 + p * 3 + 1] = R[3] * l0 + R[4] * l1 + R[5] * l2 + T[1];
    ptsg[(size_t)n * 24 + p * 3 + 2] = R[6] * l0 + R[7] * l1 + R[8] * l2 + T[2];
  }
}

// ---------------- K1: edge message GEMMs + atomic segment-sum ----------------

#define FEAT_DW 340  // 336 dwords (672 bf16) padded, 16B-aligned rows
#define H_DW 132     // 128 dwords (256 bf16) padded

__global__ __launch_bounds__(256) void k1_edge_msg(
    const int* __restrict__ eidx, const float* __restrict__ z,
    const float* __restrict__ rot, const float* __restrict__ trans,
    const u16* __restrict__ sbf, const float* __restrict__ ptsg,
    const u32* __restrict__ Wm1p, const float* __restrict__ bm1,
    const u32* __restrict__ Wm2p, const float* __restrict__ bm2,
    float* __restrict__ msg, float* __restrict__ deg) {
  __shared__ __align__(16) u32 feat[16 * FEAT_DW];
  __shared__ __align__(16) u32 hbuf[16 * H_DW];
  __shared__ int lsrc[16], ldst[16];
  const int tid = threadIdx.x, lane = tid & 31, wv = tid >> 5;
  const int e0 = blockIdx.x * 16;

  if (tid < 16) {
    int e = e0 + tid;
    lsrc[tid] = eidx[e];
    int d = eidx[N_EDGES + e];
    ldst[tid] = d;
    atomicAdd(&deg[d], 1.0f);
  }
  __syncthreads();

  // gather s[src], s[dst] (bf16 dwords, L2-resident)
  const u32* sbf_dw = reinterpret_cast<const u32*>(sbf);
  for (int idx = tid; idx < 16 * 128; idx += 256) {
    int e = idx >> 7, d = idx & 127;
    feat[e * FEAT_DW + d] = sbf_dw[(size_t)lsrc[e] * 128 + d];
    feat[e * FEAT_DW + 128 + d] = sbf_dw[(size_t)ldst[e] * 128 + d];
  }
  // z -> bf16
  for (int idx = tid; idx < 16 * 64; idx += 256) {
    int e = idx >> 6, d = idx & 63;
    const float* zr = z + (size_t)(e0 + e) * CZ;
    feat[e * FEAT_DW + 256 + d] = pack2bf(zr[2 * d], zr[2 * d + 1]);
  }
  // per-edge invariant point geometry (lanes 0..15 of wave 0 region: tid<16)
  if (tid < 16) {
    const float* R = rot + (size_t)ldst[tid] * 9;
    const float* Td = trans + (size_t)ldst[tid] * 3;
    const float* G = ptsg + (size_t)lsrc[tid] * 24;
    u16* frow = reinterpret_cast<u16*>(&feat[tid * FEAT_DW]);
    for (int p = 0; p < PP; ++p) {
      float r0 = G[p * 3 + 0] - Td[0];
      float r1 = G[p * 3 + 1] - Td[1];
      float r2 = G[p * 3 + 2] - Td[2];
      float o0 = R[0] * r0 + R[3] * r1 + R[6] * r2;  // R^T @ rel
      float o1 = R[1] * r0 + R[4] * r1 + R[7] * r2;
      float o2 = R[2] * r0 + R[5] * r1 + R[8] * r2;
      frow[640 + p * 3 + 0] = f2bf(o0);
      frow[640 + p * 3 + 1] = f2bf(o1);
      frow[640 + p * 3 + 2] = f2bf(o2);
      frow[664 + p] = f2bf(sqrtf(o0 * o0 + o1 * o1 + o2 * o2 + 1e-8f));
    }
  }
  __syncthreads();

  const int m0 = lane & 15, h = lane >> 4;
  u16* hsh = reinterpret_cast<u16*>(hbuf);

  // GEMM1: [16,672] x [672,256] -> relu -> h (bf16 in LDS)
  for (int t = 0; t < 2; ++t) {
    int nt = wv * 2 + t;
    v8f acc = {0.f, 0.f, 0.f, 0.f, 0.f, 0.f, 0.f, 0.f};
    for (int kt = 0; kt < 21; ++kt) {
      v16bf a = lds_a_frag(feat, FEAT_DW, kt, lane);
      v16bf b = glb_b_frag(Wm1p, 16, kt, nt, lane);
      acc = wmma_bf16(a, b, acc);
    }
    int n = nt * 16 + m0;
    float bias = bm1[n];
#pragma unroll
    for (int i = 0; i < 8; ++i) {
      float v = fmaxf(acc[i] + bias, 0.0f);
      hsh[(i + 8 * h) * (H_DW * 2) + n] = f2bf(v);
    }
  }
  __syncthreads();

  // GEMM2: [16,256] x [256,256] -> atomic scatter into msg[dst]
  for (int t = 0; t < 2; ++t) {
    int nt = wv * 2 + t;
    v8f acc = {0.f, 0.f, 0.f, 0.f, 0.f, 0.f, 0.f, 0.f};
    for (int kt = 0; kt < 8; ++kt) {
      v16bf a = lds_a_frag(hbuf, H_DW, kt, lane);
      v16bf b = glb_b_frag(Wm2p, 16, kt, nt, lane);
      acc = wmma_bf16(a, b, acc);
    }
    int n = nt * 16 + m0;
    float bias = bm2[n];
#pragma unroll
    for (int i = 0; i < 8; ++i) {
      int m = i + 8 * h;
      atomicAdd(&msg[(size_t)ldst[m] * CS + n], acc[i] + bias);
    }
  }
}

// ---------------- K2a: mean aggregate + residual + LN1 ----------------

__global__ __launch_bounds__(256) void k2a_ln1(
    const float* __restrict__ s, const float* __restrict__ msg,
    const float* __restrict__ deg, const float* __restrict__ mask,
    const float* __restrict__ g, const float* __restrict__ b,
    float* __restrict__ s1f, u16* __restrict__ s1b) {
  int wv = threadIdx.x >> 5, lane = threadIdx.x & 31;
  int n = blockIdx.x * 8 + wv;
  if (n >= N_NODES) return;
  float inv = 1.0f / fmaxf(deg[n], 1.0f);
  float mk = mask[n];
  float u[8];
  float sum = 0.f;
#pragma unroll
  for (int q = 0; q < 8; ++q) {
    int c = lane + 32 * q;
    float v = s[(size_t)n * CS + c] + msg[(size_t)n * CS + c] * inv * mk;
    u[q] = v;
    sum += v;
  }
  for (int o = 16; o > 0; o >>= 1) sum += __shfl_xor(sum, o);
  float mu = sum / (float)CS;
  float vs = 0.f;
#pragma unroll
  for (int q = 0; q < 8; ++q) {
    float d = u[q] - mu;
    vs += d * d;
  }
  for (int o = 16; o > 0; o >>= 1) vs += __shfl_xor(vs, o);
  float rstd = rsqrtf(vs / (float)CS + 1e-5f);
#pragma unroll
  for (int q = 0; q < 8; ++q) {
    int c = lane + 32 * q;
    float o = (u[q] - mu) * rstd * g[c] + b[c];
    s1f[(size_t)n * CS + c] = o;
    s1b[(size_t)n * CS + c] = f2bf(o);
  }
}

// ---------------- K2b: node transition MLP + LN2 + nb projection -------------

__global__ __launch_bounds__(256) void k2b_node_trans(
    const float* __restrict__ s1f, const u16* __restrict__ s1b,
    const u32* __restrict__ Wt1p, const float* __restrict__ bt1,
    const u32* __restrict__ Wt2p, const float* __restrict__ bt2,
    const u32* __restrict__ Wt3p, const float* __restrict__ bt3,
    const float* __restrict__ g2, const float* __restrict__ b2,
    const float* __restrict__ mask, const u32* __restrict__ We0p,
    const float* __restrict__ be0, float* __restrict__ sout,
    u16* __restrict__ nbb) {
  __shared__ __align__(16) u32 s1t[16 * H_DW];
  __shared__ __align__(16) u32 h1t[16 * H_DW];
  __shared__ __align__(16) u32 h2t[16 * H_DW];
  __shared__ __align__(16) u32 s2t[16 * H_DW];
  __shared__ float h3t[16 * 260];
  const int tid = threadIdx.x, lane = tid & 31, wv = tid >> 5;
  const int n0 = blockIdx.x * 16;
  const int m0 = lane & 15, h = lane >> 4;

  const u32* s1b_dw = reinterpret_cast<const u32*>(s1b);
  for (int idx = tid; idx < 16 * 128; idx += 256) {
    int r = idx >> 7, d = idx & 127;
    s1t[r * H_DW + d] = s1b_dw[(size_t)(n0 + r) * 128 + d];
  }
  __syncthreads();

  u16* h1s = reinterpret_cast<u16*>(h1t);
  u16* h2s = reinterpret_cast<u16*>(h2t);
  u16* s2s = reinterpret_cast<u16*>(s2t);

  // t1: relu(s1 @ W_t1 + b1)
  for (int t = 0; t < 2; ++t) {
    int nt = wv * 2 + t;
    v8f acc = {0.f, 0.f, 0.f, 0.f, 0.f, 0.f, 0.f, 0.f};
    for (int kt = 0; kt < 8; ++kt)
      acc = wmma_bf16(lds_a_frag(s1t, H_DW, kt, lane),
                      glb_b_frag(Wt1p, 16, kt, nt, lane), acc);
    int n = nt * 16 + m0;
    float bias = bt1[n];
#pragma unroll
    for (int i = 0; i < 8; ++i)
      h1s[(i + 8 * h) * (H_DW * 2) + n] = f2bf(fmaxf(acc[i] + bias, 0.f));
  }
  __syncthreads();

  // t2: relu(h1 @ W_t2 + b2)
  for (int t = 0; t < 2; ++t) {
    int nt = wv * 2 + t;
    v8f acc = {0.f, 0.f, 0.f, 0.f, 0.f, 0.f, 0.f, 0.f};
    for (int kt = 0; kt < 8; ++kt)
      acc = wmma_bf16(lds_a_frag(h1t, H_DW, kt, lane),
                      glb_b_frag(Wt2p, 16, kt, nt, lane), acc);
    int n = nt * 16 + m0;
    float bias = bt2[n];
#pragma unroll
    for (int i = 0; i < 8; ++i)
      h2s[(i + 8 * h) * (H_DW * 2) + n] = f2bf(fmaxf(acc[i] + bias, 0.f));
  }
  __syncthreads();

  // t3: h2 @ W_t3 + b3 (no relu), keep f32
  for (int t = 0; t < 2; ++t) {
    int nt = wv * 2 + t;
    v8f acc = {0.f, 0.f, 0.f, 0.f, 0.f, 0.f, 0.f, 0.f};
    for (int kt = 0; kt < 8; ++kt)
      acc = wmma_bf16(lds_a_frag(h2t, H_DW, kt, lane),
                      glb_b_frag(Wt3p, 16, kt, nt, lane), acc);
    int n = nt * 16 + m0;
    float bias = bt3[n];
#pragma unroll
    for (int i = 0; i < 8; ++i) h3t[(i + 8 * h) * 260 + n] = acc[i] + bias;
  }
  __syncthreads();

  // residual + LN2 + mask; stash s2 bf16 for the nb GEMM
  if (tid < 16) {
    int node = n0 + tid;
    float mk = mask[node];
    float sum = 0.f;
    for (int c = 0; c < CS; ++c) {
      float u = s1f[(size_t)node * CS + c] + h3t[tid * 260 + c];
      h3t[tid * 260 + c] = u;
      sum += u;
    }
    float mu = sum / (float)CS;
    float vs = 0.f;
    for (int c = 0; c < CS; ++c) {
      float d = h3t[tid * 260 + c] - mu;
      vs += d * d;
    }
    float rstd = rsqrtf(vs / (float)CS + 1e-5f);
    for (int c = 0; c < CS; ++c) {
      float o = ((h3t[tid * 260 + c] - mu) * rstd * g2[c] + b2[c]) * mk;
      sout[(size_t)node * CS + c] = o;
      s2s[tid * (H_DW * 2) + c] = f2bf(o);
    }
  }
  __syncthreads();

  // nb = s2 @ W_e0 + b_e0  (N=128 -> 8 n-tiles, one per wave)
  {
    int nt = wv;  // 0..7
    v8f acc = {0.f, 0.f, 0.f, 0.f, 0.f, 0.f, 0.f, 0.f};
    for (int kt = 0; kt < 8; ++kt)
      acc = wmma_bf16(lds_a_frag(s2t, H_DW, kt, lane),
                      glb_b_frag(We0p, 8, kt, nt, lane), acc);
    int n = nt * 16 + m0;
    float bias = be0[n];
#pragma unroll
    for (int i = 0; i < 8; ++i) {
      int m = i + 8 * h;
      nbb[(size_t)(n0 + m) * CZ + n] = f2bf(acc[i] + bias);
    }
  }
}

// ---------------- K3: edge transition ----------------

#define A3_DW 196  // 192 dwords (384 bf16) padded, 16B-aligned rows

__global__ __launch_bounds__(256) void k3_edge_trans(
    const int* __restrict__ eidx, const float* __restrict__ z,
    const u16* __restrict__ nbb, const u32* __restrict__ We1p,
    const float* __restrict__ be1, const u32* __restrict__ Wefp,
    const float* __restrict__ bef, const float* __restrict__ g3,
    const float* __restrict__ b3, float* __restrict__ zout) {
  __shared__ __align__(16) u32 atile[16 * A3_DW];
  __shared__ __align__(16) u32 eht[16 * A3_DW];
  __shared__ float zt[16 * H_DW];
  __shared__ int lsrc[16], ldst[16];
  const int tid = threadIdx.x, lane = tid & 31, wv = tid >> 5;
  const int e0 = blockIdx.x * 16;
  const int m0 = lane & 15, h = lane >> 4;

  if (tid < 16) {
    lsrc[tid] = eidx[e0 + tid];
    ldst[tid] = eidx[N_EDGES + e0 + tid];
  }
  __syncthreads();

  const u32* nb_dw = reinterpret_cast<const u32*>(nbb);
  for (int idx = tid; idx < 16 * 64; idx += 256) {
    int e = idx >> 6, d = idx & 63;
    atile[e * A3_DW + d] = nb_dw[(size_t)lsrc[e] * 64 + d];
    atile[e * A3_DW + 64 + d] = nb_dw[(size_t)ldst[e] * 64 + d];
    const float* zr = z + (size_t)(e0 + e) * CZ;
    atile[e * A3_DW + 128 + d] = pack2bf(zr[2 * d], zr[2 * d + 1]);
  }
  __syncthreads();

  u16* ehs = reinterpret_cast<u16*>(eht);

  // eh = relu([nb_src|nb_dst|z] @ W_e1 + b_e1)  (N=384 -> 24 tiles, 3/wave)
  for (int t = 0; t < 3; ++t) {
    int nt = wv * 3 + t;
    v8f acc = {0.f, 0.f, 0.f, 0.f, 0.f, 0.f, 0.f, 0.f};
    for (int kt = 0; kt < 12; ++kt)
      acc = wmma_bf16(lds_a_frag(atile, A3_DW, kt, lane),
                      glb_b_frag(We1p, 24, kt, nt, lane), acc);
    int n = nt * 16 + m0;
    float bias = be1[n];
#pragma unroll
    for (int i = 0; i < 8; ++i)
      ehs[(i + 8 * h) * (A3_DW * 2) + n] = f2bf(fmaxf(acc[i] + bias, 0.f));
  }
  __syncthreads();

  // z_mid = eh @ W_ef + b_ef  (N=128 -> 8 tiles, 1/wave)
  {
    int nt = wv;
    v8f acc = {0.f, 0.f, 0.f, 0.f, 0.f, 0.f, 0.f, 0.f};
    for (int kt = 0; kt < 12; ++kt)
      acc = wmma_bf16(lds_a_frag(eht, A3_DW, kt, lane),
                      glb_b_frag(Wefp, 8, kt, nt, lane), acc);
    int n = nt * 16 + m0;
    float bias = bef[n];
#pragma unroll
    for (int i = 0; i < 8; ++i) zt[(i + 8 * h) * H_DW + n] = acc[i] + bias;
  }
  __syncthreads();

  // LN3 per edge row, write z_out
  if (tid < 16) {
    float sum = 0.f;
    for (int c = 0; c < CZ; ++c) sum += zt[tid * H_DW + c];
    float mu = sum / (float)CZ;
    float vs = 0.f;
    for (int c = 0; c < CZ; ++c) {
      float d = zt[tid * H_DW + c] - mu;
      vs += d * d;
    }
    float rstd = rsqrtf(vs / (float)CZ + 1e-5f);
    for (int c = 0; c < CZ; ++c) {
      float o = (zt[tid * H_DW + c] - mu) * rstd * g3[c] + b3[c];
      zout[(size_t)(e0 + tid) * CZ + c] = o;
    }
  }
}

// ---------------- workspace layout ----------------

constexpr size_t OFF_SBF = 0;
constexpr size_t OFF_PTS = OFF_SBF + (size_t)N_NODES * CS * 2;       // s bf16
constexpr size_t OFF_MSG = OFF_PTS + (size_t)N_NODES * 24 * 4;       // pts_glob
constexpr size_t OFF_DEG = OFF_MSG + (size_t)N_NODES * CS * 4;       // msg f32
constexpr size_t OFF_S1F = OFF_DEG + 64 * 1024;                      // deg (padded)
constexpr size_t OFF_S1B = OFF_S1F + (size_t)N_NODES * CS * 4;
constexpr size_t OFF_NB  = OFF_S1B + (size_t)N_NODES * CS * 2;
constexpr size_t OFF_Wm1 = OFF_NB + (size_t)N_NODES * CZ * 2;
constexpr size_t OFF_Wm2 = OFF_Wm1 + (size_t)DMSG * CH * 2;
constexpr size_t OFF_Wt1 = OFF_Wm2 + (size_t)CH * CS * 2;
constexpr size_t OFF_Wt2 = OFF_Wt1 + (size_t)CS * CS * 2;
constexpr size_t OFF_Wt3 = OFF_Wt2 + (size_t)CS * CS * 2;
constexpr size_t OFF_We0 = OFF_Wt3 + (size_t)CS * CS * 2;
constexpr size_t OFF_We1 = OFF_We0 + (size_t)CS * (CS / 2) * 2;
constexpr size_t OFF_Wef = OFF_We1 + (size_t)DEH * DEH * 2;

extern "C" void kernel_launch(void* const* d_in, const int* in_sizes, int n_in,
                              void* d_out, int out_size, void* d_ws, size_t ws_size,
                              hipStream_t stream) {
  (void)in_sizes; (void)n_in; (void)out_size; (void)ws_size;
  const float* s     = (const float*)d_in[0];
  const float* z     = (const float*)d_in[1];
  const int*   eidx  = (const int*)d_in[2];
  const float* rot   = (const float*)d_in[3];
  const float* trans = (const float*)d_in[4];
  const float* mask  = (const float*)d_in[5];
  const float* Wpts  = (const float*)d_in[6];
  const float* bpts  = (const float*)d_in[7];
  const float* Wm1   = (const float*)d_in[8];
  const float* bm1   = (const float*)d_in[9];
  const float* Wm2   = (const float*)d_in[10];
  const float* bm2   = (const float*)d_in[11];
  const float* g1    = (const float*)d_in[12];
  const float* b1    = (const float*)d_in[13];
  const float* Wt1   = (const float*)d_in[14];
  const float* bt1   = (const float*)d_in[15];
  const float* Wt2   = (const float*)d_in[16];
  const float* bt2   = (const float*)d_in[17];
  const float* Wt3   = (const float*)d_in[18];
  const float* bt3   = (const float*)d_in[19];
  const float* g2    = (const float*)d_in[20];
  const float* b2    = (const float*)d_in[21];
  const float* We0   = (const float*)d_in[22];
  const float* be0   = (const float*)d_in[23];
  const float* We1   = (const float*)d_in[24];
  const float* be1   = (const float*)d_in[25];
  const float* Wef   = (const float*)d_in[26];
  const float* bef   = (const float*)d_in[27];
  const float* g3    = (const float*)d_in[28];
  const float* b3    = (const float*)d_in[29];

  char* ws = (char*)d_ws;
  u16*  sbf  = (u16*)(ws + OFF_SBF);
  float* ptsg = (float*)(ws + OFF_PTS);
  float* msg  = (float*)(ws + OFF_MSG);
  float* deg  = (float*)(ws + OFF_DEG);
  float* s1f  = (float*)(ws + OFF_S1F);
  u16*  s1b  = (u16*)(ws + OFF_S1B);
  u16*  nbb  = (u16*)(ws + OFF_NB);
  u32*  Wm1p = (u32*)(ws + OFF_Wm1);
  u32*  Wm2p = (u32*)(ws + OFF_Wm2);
  u32*  Wt1p = (u32*)(ws + OFF_Wt1);
  u32*  Wt2p = (u32*)(ws + OFF_Wt2);
  u32*  Wt3p = (u32*)(ws + OFF_Wt3);
  u32*  We0p = (u32*)(ws + OFF_We0);
  u32*  We1p = (u32*)(ws + OFF_We1);
  u32*  Wefp = (u32*)(ws + OFF_Wef);

  float* sout = (float*)d_out;
  float* zout = sout + (size_t)N_NODES * CS;

  // zero msg + deg every call (graph-replay safe)
  long long zn = (long long)(OFF_S1F - OFF_MSG) / 4;
  k_zero_f32<<<2048, 256, 0, stream>>>(msg, zn);

  // weight packing (small; L2-resident thereafter)
  k_pack_b<<<256, 256, 0, stream>>>(Wm1, DMSG, CH, Wm1p);
  k_pack_b<<<256, 256, 0, stream>>>(Wm2, CH, CS, Wm2p);
  k_pack_b<<<256, 256, 0, stream>>>(Wt1, CS, CS, Wt1p);
  k_pack_b<<<256, 256, 0, stream>>>(Wt2, CS, CS, Wt2p);
  k_pack_b<<<256, 256, 0, stream>>>(Wt3, CS, CS, Wt3p);
  k_pack_b<<<256, 256, 0, stream>>>(We0, CS, CS / 2, We0p);
  k_pack_b<<<256, 256, 0, stream>>>(We1, DEH, DEH, We1p);
  k_pack_b<<<256, 256, 0, stream>>>(Wef, DEH, CZ, Wefp);

  k0_node_prep<<<N_NODES / 8, 256, 0, stream>>>(s, rot, trans, Wpts, bpts, sbf, ptsg);
  k1_edge_msg<<<N_EDGES / 16, 256, 0, stream>>>(eidx, z, rot, trans, sbf, ptsg,
                                                Wm1p, bm1, Wm2p, bm2, msg, deg);
  k2a_ln1<<<N_NODES / 8, 256, 0, stream>>>(s, msg, deg, mask, g1, b1, s1f, s1b);
  k2b_node_trans<<<N_NODES / 16, 256, 0, stream>>>(s1f, s1b, Wt1p, bt1, Wt2p, bt2,
                                                   Wt3p, bt3, g2, b2, mask, We0p,
                                                   be0, sout, nbb);
  k3_edge_trans<<<N_EDGES / 16, 256, 0, stream>>>(eidx, z, nbb, We1p, be1, Wefp,
                                                  bef, g3, b3, zout);
}
